// SentimentModelNoEmbed_28321014349948
// MI455X (gfx1250) — compile-verified, matrix operand order checked by reading, not run on hardware
//
#include <hip/hip_runtime.h>
#include <hip/hip_bf16.h>

// ---------------------------------------------------------------------------
// Types for WMMA fragments (CDNA5 gfx1250, wave32)
// ---------------------------------------------------------------------------
typedef __attribute__((ext_vector_type(16))) __bf16       v16bf;
typedef __attribute__((ext_vector_type(8)))  float        v8f;
typedef __attribute__((ext_vector_type(4)))  unsigned int u32x4;

union Frag {
  u32x4 q[2];
  v16bf v;
};

__device__ __forceinline__ unsigned short f2bf(float f) {
  unsigned u = __builtin_bit_cast(unsigned, f);
  u += 0x7FFFu + ((u >> 16) & 1u);           // round-to-nearest-even
  return (unsigned short)(u >> 16);
}
__device__ __forceinline__ float sigmoid_(float x) {
  return 1.0f / (1.0f + __expf(-x));
}
__device__ __forceinline__ float tanh_(float x) {
  float e = __expf(2.0f * x);                // inf-safe: e=inf -> 1, e=0 -> -1
  return 1.0f - 2.0f / (e + 1.0f);
}

// ---------------------------------------------------------------------------
// Utility kernels
// ---------------------------------------------------------------------------
__global__ void k_f32_to_bf16(const float* __restrict__ src,
                              unsigned short* __restrict__ dst, int n) {
  int i = blockIdx.x * blockDim.x + threadIdx.x;
  int stride = gridDim.x * blockDim.x;
  for (; i < n; i += stride) dst[i] = f2bf(src[i]);
}

__global__ void k_bias(const float* __restrict__ a, const float* __restrict__ b,
                       float* __restrict__ o) {
  int i = blockIdx.x * blockDim.x + threadIdx.x;
  if (i < 2048) o[i] = a[i] + b[i];
}

__global__ void k_zero(float* __restrict__ p, int n) {
  int i = blockIdx.x * blockDim.x + threadIdx.x;
  if (i < n) p[i] = 0.0f;
}

// Pack W [N=2048, K] (row-major fp32, gate order i,f,g,o along N) into the
// per-lane WMMA B-fragment layout for v_wmma_f32_16x16x32_bf16.
// Fragment (ntile, kb): 32 lanes x 16 bf16 (32 B per lane, contiguous).
// Lane l<16  -> N = ntile*16+l,      K = kb*32 + {0..7, 16..23}
// Lane l>=16 -> N = ntile*16+(l-16), K = kb*32 + {8..15, 24..31}
__global__ void k_pack_w(const float* __restrict__ W,
                         unsigned int* __restrict__ dst, int K) {
  int nKB = K >> 5;
  int total = 128 * nKB * 32;
  int gid = blockIdx.x * blockDim.x + threadIdx.x;
  if (gid >= total) return;
  int lane  = gid & 31;
  int kb    = (gid >> 5) % nKB;
  int ntile = gid / (32 * nKB);
  int n     = ntile * 16 + (lane & 15);
  int kbase = kb * 32 + ((lane & 16) ? 8 : 0);
  unsigned int out[8];
#pragma unroll
  for (int j = 0; j < 8; ++j) {
    int e0 = 2 * j, e1 = 2 * j + 1;
    int k0 = kbase + e0 + ((e0 >= 8) ? 8 : 0);
    int k1 = kbase + e1 + ((e1 >= 8) ? 8 : 0);
    unsigned lo = f2bf(W[(size_t)n * K + k0]);
    unsigned hi = f2bf(W[(size_t)n * K + k1]);
    out[j] = lo | (hi << 16);
  }
  unsigned int* p = dst + (size_t)gid * 8;
  *(u32x4*)p       = *(u32x4*)&out[0];
  *(u32x4*)(p + 4) = *(u32x4*)&out[4];
}

// ---------------------------------------------------------------------------
// Phase A: xg[r, n] = A[r, :K] @ W^T + bias  for a chunk of rows (r = b*TC+t)
// A is bf16 row-major; W is pre-packed fragments; output fp32 [rows, 2048].
// Each wave: one 16-row tile x one 128-col strip (8 n-tiles), full K.
// Grid: 1024 blocks x 256 threads = 8192 waves = 512 rowtiles * 16 strips.
// A fragments register-cached (single pass, no LICM/spill hazard); latency
// hidden by the sheer wave count.
// ---------------------------------------------------------------------------
template <int NKB>
__global__ __launch_bounds__(256) void k_xg_gemm(
    const unsigned short* __restrict__ Abf, const unsigned int* __restrict__ wpack,
    const float* __restrict__ bias, float* __restrict__ xg,
    int a_b_stride, int a_t0, int TC) {
  const int K = NKB * 32;
  int lane = threadIdx.x & 31;
  int wid  = blockIdx.x * 8 + (threadIdx.x >> 5);
  int rowtile = wid >> 4;
  int strip   = wid & 15;
  int lr    = lane & 15;
  int khalf = (lane & 16) ? 8 : 0;

  int flat = rowtile * 16 + lr;           // chunk-flat row of this lane's A row
  int b = flat / TC, tl = flat - b * TC;  // TC multiple of 16 -> no b straddle
  const unsigned short* arow = Abf + (size_t)(b * a_b_stride + a_t0 + tl) * K;

  Frag A[NKB];
#pragma unroll
  for (int kb = 0; kb < NKB; ++kb) {
    int k0 = kb * 32 + khalf;
    A[kb].q[0] = *(const u32x4*)(arow + k0);
    A[kb].q[1] = *(const u32x4*)(arow + k0 + 16);
  }

#pragma unroll
  for (int nt = 0; nt < 8; ++nt) {
    int ntile = strip * 8 + nt;
    v8f acc = {};
    const unsigned int* wp = wpack + ((size_t)(ntile * NKB) * 32 + lane) * 8;
#pragma unroll
    for (int kb = 0; kb < NKB; ++kb) {
      Frag B;
      const unsigned int* p = wp + (size_t)kb * 32 * 8;
      B.q[0] = *(const u32x4*)p;
      B.q[1] = *(const u32x4*)(p + 4);
      acc = __builtin_amdgcn_wmma_f32_16x16x32_bf16(
          false, A[kb].v, false, B.v, (short)0, acc, false, false);
    }
    float bv = bias[ntile * 16 + lr];
    // D layout: VGPR r, lane l -> row = r + 8*(l>>4), col = l&15
    float* out = xg + (size_t)(rowtile * 16 + khalf) * 2048 + ntile * 16 + lr;
#pragma unroll
    for (int r = 0; r < 8; ++r) out[(size_t)r * 2048] = acc[r] + bv;
  }
}

// ---------------------------------------------------------------------------
// Phase B: recurrence for one chunk of TC timesteps.
// 8 workgroups (one per 16-row batch tile) x 16 waves (512 threads, 4 waves
// per SIMD32 for SMT latency hiding). h kept in LDS (bf16, padded rows:
// stride 520 shorts => conflict-free b128 reads), c kept in registers.
// Each wave owns 2 hidden-column tiles; per step: 2 tiles x 4 gates x 16 kb
// = 128 v_wmma_f32_16x16x32_bf16 + fused sigmoid/tanh.
//
// NOTE: `wpack` and `h_out_bf` intentionally NOT __restrict__: the in-loop
// stores through h_out_bf may-alias the weight loads, which prevents LLVM
// from hoisting all weight fragments out of the t-loop and spilling them to
// scratch (observed in an earlier build). Weights re-stream from L2 (2 MB,
// resident in the 192 MB L2) every step, as intended.
//
// Memory-level parallelism: per k-block, all 4 gate B-fragments are loaded
// into Bc[4] as one burst (8x global_load_b128 in flight), then consumed
// in order by the 4 WMMAs -> partial s_wait_loadcnt instead of a full stall
// per fragment. Register budget ~190 VGPRs: no MSB mode, no spills.
// ---------------------------------------------------------------------------
__global__ __launch_bounds__(512) void k_lstm_rec(
    const float* __restrict__ xg,            // [128, TC, 2048] fp32 (xg+bias)
    const unsigned int* wpack,               // packed W_hh, NKB=16 (no restrict!)
    float* __restrict__ h_state,             // [128,512] fp32 carry
    float* __restrict__ c_state,             // [128,512] fp32 carry
    unsigned short* h_out_bf,                // [128, TC, 512] bf16 or null
    float* __restrict__ h_last,              // [128,512] fp32 (written if flag)
    int TC, int write_last) {
  __shared__ unsigned short h_sh[16][520];
  const int tid  = threadIdx.x;
  const int lane = tid & 31;
  const int wave = tid >> 5;                 // 0..15
  const int wg   = blockIdx.x;               // batch rows [wg*16, wg*16+16)
  const int lr    = lane & 15;
  const int khalf = (lane & 16) ? 8 : 0;     // K-half select == C/D row offset

  // h carry -> LDS (bf16)
  for (int i = tid; i < 16 * 512; i += 512)
    h_sh[i >> 9][i & 511] =
        f2bf(h_state[(size_t)(wg * 16 + (i >> 9)) * 512 + (i & 511)]);

  // c carry -> registers (2 column tiles per wave)
  v8f creg[2];
#pragma unroll
  for (int q = 0; q < 2; ++q) {
    int col = (wave * 2 + q) * 16 + lr;
#pragma unroll
    for (int r = 0; r < 8; ++r)
      creg[q][r] = c_state[(size_t)(wg * 16 + khalf + r) * 512 + col];
  }
  __syncthreads();

  for (int t = 0; t < TC; ++t) {
    unsigned short hreg[2][8];               // buffered h(t) (bf16 bits)

#pragma unroll
    for (int q = 0; q < 2; ++q) {
      int ct   = wave * 2 + q;
      int col0 = ct * 16;
      // init accumulators from precomputed xg (already includes both biases)
      v8f acc[4];
      const float* xp =
          xg + ((size_t)(wg * 16 + khalf) * TC + t) * 2048 + col0 + lr;
#pragma unroll
      for (int g = 0; g < 4; ++g)
#pragma unroll
        for (int r = 0; r < 8; ++r)
          acc[g][r] = xp[(size_t)r * TC * 2048 + g * 512];

#pragma unroll
      for (int kb = 0; kb < 16; ++kb) {
        // A fragment straight from LDS (shared by the 4 gate WMMAs)
        Frag Af;
        int k0 = kb * 32 + khalf;
        Af.q[0] = *(const u32x4*)&h_sh[lr][k0];
        Af.q[1] = *(const u32x4*)&h_sh[lr][k0 + 16];
        // Burst-load all 4 gate fragments for this k-block (8x b128 in
        // flight), then consume in order -> partial loadcnt waits.
        Frag Bc[4];
#pragma unroll
        for (int g = 0; g < 4; ++g) {
          const unsigned int* p =
              wpack + ((size_t)((g * 32 + ct) * 16 + kb) * 32 + lane) * 8;
          Bc[g].q[0] = *(const u32x4*)p;
          Bc[g].q[1] = *(const u32x4*)(p + 4);
        }
#pragma unroll
        for (int g = 0; g < 4; ++g)
          acc[g] = __builtin_amdgcn_wmma_f32_16x16x32_bf16(
              false, Af.v, false, Bc[g].v, (short)0, acc[g], false, false);
      }

      // gate math + state update (i,f,g,o order); h_sh NOT written here
#pragma unroll
      for (int r = 0; r < 8; ++r) {
        float iv = sigmoid_(acc[0][r]);
        float fv = sigmoid_(acc[1][r]);
        float gv = tanh_(acc[2][r]);
        float ov = sigmoid_(acc[3][r]);
        float c  = fv * creg[q][r] + iv * gv;
        creg[q][r] = c;
        float h = ov * tanh_(c);
        int row = khalf + r;
        unsigned short hb = f2bf(h);
        hreg[q][r] = hb;
        if (h_out_bf)
          h_out_bf[((size_t)(wg * 16 + row) * TC + t) * 512 + col0 + lr] = hb;
        if (t == TC - 1) {
          h_state[(size_t)(wg * 16 + row) * 512 + col0 + lr] = h;
          if (write_last)
            h_last[(size_t)(wg * 16 + row) * 512 + col0 + lr] = h;
        }
      }
    }

    __syncthreads();  // all reads of h(t-1) from h_sh complete
#pragma unroll
    for (int q = 0; q < 2; ++q) {
      int col = (wave * 2 + q) * 16 + lr;
#pragma unroll
      for (int r = 0; r < 8; ++r) h_sh[khalf + r][col] = hreg[q][r];
    }
    __syncthreads();  // h(t) visible to all waves for next step
  }

  // c carry back
#pragma unroll
  for (int q = 0; q < 2; ++q) {
    int col = (wave * 2 + q) * 16 + lr;
#pragma unroll
    for (int r = 0; r < 8; ++r)
      c_state[(size_t)(wg * 16 + khalf + r) * 512 + col] = creg[q][r];
  }
}

// ---------------------------------------------------------------------------
// Final FC + sigmoid: out[b] = sigmoid(h_last[b,:] . w_fc + b_fc)
// ---------------------------------------------------------------------------
__global__ void k_fc(const float* __restrict__ hl, const float* __restrict__ w,
                     const float* __restrict__ bfc, float* __restrict__ out) {
  int i = blockIdx.x * blockDim.x + threadIdx.x;
  if (i < 128) {
    float s = bfc[0];
    for (int k = 0; k < 512; ++k) s += hl[(size_t)i * 512 + k] * w[k];
    out[i] = sigmoid_(s);
  }
}

// ---------------------------------------------------------------------------
// Launch: B=128, T=512, E=256, H=512. T chunked into 8 x TC=64.
// ---------------------------------------------------------------------------
extern "C" void kernel_launch(void* const* d_in, const int* in_sizes, int n_in,
                              void* d_out, int out_size, void* d_ws,
                              size_t ws_size, hipStream_t stream) {
  (void)in_sizes; (void)n_in; (void)out_size; (void)ws_size;
  const float* x     = (const float*)d_in[0];
  const float* w_ih0 = (const float*)d_in[1];
  const float* w_hh0 = (const float*)d_in[2];
  const float* b_ih0 = (const float*)d_in[3];
  const float* b_hh0 = (const float*)d_in[4];
  const float* w_ih1 = (const float*)d_in[5];
  const float* w_hh1 = (const float*)d_in[6];
  const float* b_ih1 = (const float*)d_in[7];
  const float* b_hh1 = (const float*)d_in[8];
  const float* w_fc  = (const float*)d_in[9];
  const float* b_fc  = (const float*)d_in[10];
  float* out = (float*)d_out;

  const int TC = 64;
  char* ws = (char*)d_ws;
  size_t off = 0;
  auto alloc = [&](size_t bytes) {
    char* p = ws + off;
    off = (off + bytes + 255) & ~(size_t)255;
    return p;
  };
  unsigned short* x_bf  = (unsigned short*)alloc((size_t)128 * 512 * 256 * 2);
  unsigned int*   wih0p = (unsigned int*)alloc((size_t)128 * 8 * 32 * 32);
  unsigned int*   whh0p = (unsigned int*)alloc((size_t)128 * 16 * 32 * 32);
  unsigned int*   wih1p = (unsigned int*)alloc((size_t)128 * 16 * 32 * 32);
  unsigned int*   whh1p = (unsigned int*)alloc((size_t)128 * 16 * 32 * 32);
  float*          bias0 = (float*)alloc(2048 * 4);
  float*          bias1 = (float*)alloc(2048 * 4);
  float*          xgbuf = (float*)alloc((size_t)128 * TC * 2048 * 4);
  unsigned short* h1c   = (unsigned short*)alloc((size_t)128 * TC * 512 * 2);
  float*          hs0   = (float*)alloc((size_t)128 * 512 * 4);
  float*          cs0   = (float*)alloc((size_t)128 * 512 * 4);
  float*          hs1   = (float*)alloc((size_t)128 * 512 * 4);
  float*          cs1   = (float*)alloc((size_t)128 * 512 * 4);
  float*          hlast = (float*)alloc((size_t)128 * 512 * 4);

  // Prep: bf16 conversion, weight packing, bias fusion, state zeroing.
  k_f32_to_bf16<<<4096, 256, 0, stream>>>(x, x_bf, 128 * 512 * 256);
  k_pack_w<<<(128 * 8 * 32) / 256, 256, 0, stream>>>(w_ih0, wih0p, 256);
  k_pack_w<<<(128 * 16 * 32) / 256, 256, 0, stream>>>(w_hh0, whh0p, 512);
  k_pack_w<<<(128 * 16 * 32) / 256, 256, 0, stream>>>(w_ih1, wih1p, 512);
  k_pack_w<<<(128 * 16 * 32) / 256, 256, 0, stream>>>(w_hh1, whh1p, 512);
  k_bias<<<8, 256, 0, stream>>>(b_ih0, b_hh0, bias0);
  k_bias<<<8, 256, 0, stream>>>(b_ih1, b_hh1, bias1);
  k_zero<<<256, 256, 0, stream>>>(hs0, 128 * 512);
  k_zero<<<256, 256, 0, stream>>>(cs0, 128 * 512);
  k_zero<<<256, 256, 0, stream>>>(hs1, 128 * 512);
  k_zero<<<256, 256, 0, stream>>>(cs1, 128 * 512);

  for (int ch = 0; ch < 8; ++ch) {
    // Layer 0: input GEMM for chunk (A = x, row = b*512 + ch*TC + t)
    k_xg_gemm<8><<<1024, 256, 0, stream>>>(x_bf, wih0p, bias0, xgbuf,
                                           512, ch * TC, TC);
    k_lstm_rec<<<8, 512, 0, stream>>>(xgbuf, whh0p, hs0, cs0, h1c, hlast, TC, 0);
    // Layer 1: input GEMM over layer-0 chunk output (A = h1c, row = b*TC + t)
    k_xg_gemm<16><<<1024, 256, 0, stream>>>(h1c, wih1p, bias1, xgbuf,
                                            TC, 0, TC);
    k_lstm_rec<<<8, 512, 0, stream>>>(xgbuf, whh1p, hs1, cs1, nullptr, hlast,
                                      TC, (ch == 7) ? 1 : 0);
  }
  k_fc<<<1, 128, 0, stream>>>(hlast, w_fc, b_fc, out);
}
